// CrossAttentionLayer_7481833029923
// MI455X (gfx1250) — matrix operand order repaired and checked by reading, not compiled, not run on hardware
//
#include <hip/hip_runtime.h>
#include <hip/hip_bf16.h>
#include <stdint.h>

typedef _Float16 half_t;
typedef __attribute__((ext_vector_type(16))) _Float16 v16h;
typedef __attribute__((ext_vector_type(8)))  _Float16 v8h;
typedef __attribute__((ext_vector_type(8)))  float    v8f;

#define B_SZ   4
#define SEQ    2048
#define DMODEL 1024
#define NHEADS 16
#define DHEAD  64
#define HIDDEN 4096
#define ROWS   (B_SZ * SEQ)   // 8192

// ---------------------------------------------------------------------------
// CDNA5 async global->LDS copy (ASYNCcnt-tracked), ISA §15.18.3 / §10 (async).
// LDS operand is the low 32 bits of the generic pointer (aperture rule).
// ---------------------------------------------------------------------------
__device__ __forceinline__ void async_copy_b128(void* lds_ptr, const void* gptr) {
    const uint32_t lds = (uint32_t)(uintptr_t)lds_ptr;
    const uint64_t ga  = (uint64_t)(uintptr_t)gptr;
    asm volatile("global_load_async_to_lds_b128 %0, %1, off"
                 :: "v"(lds), "v"(ga) : "memory");
}
__device__ __forceinline__ void wait_async0() {
    asm volatile("s_wait_asynccnt 0" ::: "memory");
}
__device__ __forceinline__ void wait_async4() {
    asm volatile("s_wait_asynccnt 4" ::: "memory");
}

// ---------------------------------------------------------------------------
// f32 [K x N] -> f16 transposed [N x K]  (so GEMM B-fragments read contiguous K)
// ---------------------------------------------------------------------------
__global__ void transpose_to_f16(const float* __restrict__ W, half_t* __restrict__ Wt,
                                 int K, int N) {
    __shared__ float tile[32][33];
    const int n0 = blockIdx.x * 32, k0 = blockIdx.y * 32;
    const int tx = threadIdx.x, ty = threadIdx.y;
#pragma unroll
    for (int i = 0; i < 4; ++i)
        tile[ty + i * 8][tx] = W[(size_t)(k0 + ty + i * 8) * N + n0 + tx];
    __syncthreads();
#pragma unroll
    for (int i = 0; i < 4; ++i)
        Wt[(size_t)(n0 + ty + i * 8) * K + k0 + tx] = (half_t)tile[tx][ty + i * 8];
}

// ---------------------------------------------------------------------------
// LayerNorm (D=1024) -> f16
// ---------------------------------------------------------------------------
__global__ void layernorm_f16(const float* __restrict__ X, const float* __restrict__ g,
                              const float* __restrict__ b, half_t* __restrict__ out) {
    const int row = blockIdx.x;
    const float* x = X + (size_t)row * DMODEL;
    float v[4], s = 0.f, ss = 0.f;
#pragma unroll
    for (int i = 0; i < 4; ++i) {
        v[i] = x[threadIdx.x + i * 256];
        s += v[i]; ss += v[i] * v[i];
    }
    __shared__ float rs[256], rss[256];
    rs[threadIdx.x] = s; rss[threadIdx.x] = ss;
    __syncthreads();
    for (int off = 128; off > 0; off >>= 1) {
        if (threadIdx.x < off) {
            rs[threadIdx.x]  += rs[threadIdx.x + off];
            rss[threadIdx.x] += rss[threadIdx.x + off];
        }
        __syncthreads();
    }
    const float mu  = rs[0] * (1.f / DMODEL);
    const float var = rss[0] * (1.f / DMODEL) - mu * mu;
    const float inv = rsqrtf(var + 1e-5f);
    half_t* o = out + (size_t)row * DMODEL;
#pragma unroll
    for (int i = 0; i < 4; ++i) {
        const int c = threadIdx.x + i * 256;
        o[c] = (half_t)((v[i] - mu) * inv * g[c] + b[c]);
    }
}

// ---------------------------------------------------------------------------
// WMMA GEMM: C[R x N] = A[R x K] (f16, row-major) @ Bt[N x K]^T (f16, B^T row-major)
// Block tile 128x128, 8 waves (2x4), wave tile 64x32 (4x2 frags), K-step 32.
// Double-buffered LDS fed by global_load_async_to_lds_b128 (ASYNCcnt pipeline).
// EPI 0: Ch = f16(acc * scale)
// EPI 1: Cf = acc + bias[n] + resid[m,n]          (f32 out)
// EPI 2: Ch = f16(silu(acc + bias[n]))
// ---------------------------------------------------------------------------
template <int EPI>
__global__ __launch_bounds__(256, 2) void gemm_wmma(
    const half_t* __restrict__ A, const half_t* __restrict__ Bt,
    const float* __restrict__ bias, const float* __restrict__ resid,
    float* __restrict__ Cf, half_t* __restrict__ Ch,
    int Ncols, int K, float scale)
{
    __shared__ half_t sA[2][128][40];   // 32-wide K tile, pad to 40 halves (80B rows)
    __shared__ half_t sB[2][128][40];

    const int tid  = threadIdx.x;
    const int lane = tid & 31, wv = tid >> 5;
    const int m16  = lane & 15, grp = lane >> 4;
    const int wm   = wv >> 2,  wn = wv & 3;
    const size_t rowBase = (size_t)blockIdx.y * 128;
    const size_t colBase = (size_t)blockIdx.x * 128;

    // each thread owns two 8-half chunks of each tile (512 chunks total)
    int cr[2], co[2];
    cr[0] = tid >> 2;         co[0] = (tid & 3) * 8;
    cr[1] = (tid + 256) >> 2; co[1] = ((tid + 256) & 3) * 8;

    auto issue = [&](int buf, int k0) {
#pragma unroll
        for (int i = 0; i < 2; ++i) {
            async_copy_b128(&sA[buf][cr[i]][co[i]], A + (rowBase + cr[i]) * K + k0 + co[i]);
            async_copy_b128(&sB[buf][cr[i]][co[i]], Bt + (colBase + cr[i]) * K + k0 + co[i]);
        }
    };

    v8f acc[4][2];
#pragma unroll
    for (int i = 0; i < 4; ++i)
#pragma unroll
        for (int j = 0; j < 2; ++j)
#pragma unroll
            for (int e = 0; e < 8; ++e) acc[i][j][e] = 0.f;

    const int nT = K / 32;
    issue(0, 0);
    for (int t = 0; t < nT; ++t) {
        const int buf = t & 1;
        if (t + 1 < nT) {        // prefetch next tile into the other buffer
            issue(buf ^ 1, (t + 1) * 32);
            wait_async4();       // retire exactly tile t's 4 in-order async ops
        } else {
            wait_async0();
        }
        __syncthreads();

        // ---- fragments (ISA 16-bit A/B lane layouts)
        union { v16h v; v8h h[2]; } af[4], bf[2];
#pragma unroll
        for (int fm = 0; fm < 4; ++fm) {
            const int r = wm * 64 + fm * 16 + m16;
            af[fm].h[0] = *(const v8h*)&sA[buf][r][grp * 8];        // K = grp*8 .. +7
            af[fm].h[1] = *(const v8h*)&sA[buf][r][16 + grp * 8];   // K = 16+grp*8 .. +7
        }
#pragma unroll
        for (int fn = 0; fn < 2; ++fn) {
            const int r = wn * 32 + fn * 16 + m16;
            bf[fn].h[0] = *(const v8h*)&sB[buf][r][grp * 8];        // K = grp*8 .. +7
            bf[fn].h[1] = *(const v8h*)&sB[buf][r][16 + grp * 8];   // K = 16+grp*8 .. +7
        }
#pragma unroll
        for (int fm = 0; fm < 4; ++fm)
#pragma unroll
            for (int fn = 0; fn < 2; ++fn)
                acc[fm][fn] = __builtin_amdgcn_wmma_f32_16x16x32_f16(
                    false, af[fm].v, false, bf[fn].v, (short)0, acc[fm][fn], false, false);
        __syncthreads();   // all waves done reading buf before it is refilled
    }

    // ---- epilogue (C layout: row = grp*8 + r, col = m16)
#pragma unroll
    for (int fm = 0; fm < 4; ++fm)
#pragma unroll
        for (int fn = 0; fn < 2; ++fn)
#pragma unroll
            for (int r = 0; r < 8; ++r) {
                const size_t gm = rowBase + wm * 64 + fm * 16 + grp * 8 + r;
                const size_t gn = colBase + wn * 32 + fn * 16 + m16;
                const float vv = acc[fm][fn][r];
                if (EPI == 0) {
                    Ch[gm * Ncols + gn] = (half_t)(vv * scale);
                } else if (EPI == 1) {
                    Cf[gm * Ncols + gn] = vv + bias[gn] + resid[gm * Ncols + gn];
                } else {
                    const float t = vv + bias[gn];
                    Ch[gm * Ncols + gn] = (half_t)(t / (1.f + __expf(-t)));
                }
            }
}

// ---------------------------------------------------------------------------
// Flash attention: grid (SEQ/128, B*H); 8 waves/block, 16 query rows per wave.
// Q pre-scaled by 1/sqrt(64). Online softmax; QK^T and P@V via WMMA.
// K tile staged via async global->LDS; V staged through VGPRs (transpose scatter).
// ---------------------------------------------------------------------------
__global__ __launch_bounds__(256, 2) void attention_wmma(
    const half_t* __restrict__ Q, const half_t* __restrict__ Kh,
    const half_t* __restrict__ V, half_t* __restrict__ O)
{
    __shared__ half_t sK [64][72];      // raw K tile  (j, d)
    __shared__ half_t sVt[64][72];      // V transposed (d, j)
    __shared__ half_t sP [8][16][72];   // per-wave P strip (m, j)

    const int tid  = threadIdx.x;
    const int lane = tid & 31, wv = tid >> 5;
    const int m16  = lane & 15, grp = lane >> 4;
    const int b = blockIdx.y >> 4, h = blockIdx.y & 15;
    const int qRow0 = blockIdx.x * 128 + wv * 16;
    const size_t base = ((size_t)b * SEQ) * DMODEL + (size_t)h * DHEAD;

    // Q strip -> two A-fragments (K-chunks d=[0,32),[32,64)), straight from global
    union { v16h v; v8h h2[2]; } qf[2];
    {
        const half_t* qp = Q + base + (size_t)(qRow0 + m16) * DMODEL;
#pragma unroll
        for (int kc = 0; kc < 2; ++kc) {
            qf[kc].h2[0] = *(const v8h*)(qp + kc * 32 + grp * 8);
            qf[kc].h2[1] = *(const v8h*)(qp + kc * 32 + 16 + grp * 8);
        }
    }

    float mrow[8], lrow[8];
    v8f Ofr[4];
#pragma unroll
    for (int r = 0; r < 8; ++r) { mrow[r] = -1e30f; lrow[r] = 0.f; }
#pragma unroll
    for (int fd = 0; fd < 4; ++fd)
#pragma unroll
        for (int e = 0; e < 8; ++e) Ofr[fd][e] = 0.f;

    for (int j0 = 0; j0 < SEQ; j0 += 64) {
        // ---- stage K (async direct to LDS) and V (transposed via VGPRs)
#pragma unroll
        for (int i = 0; i < 2; ++i) {
            const int c = tid + i * 256;           // 512 chunks of 8 halves
            const int r = c >> 3, off = (c & 7) * 8;
            const size_t g = base + (size_t)(j0 + r) * DMODEL + off;
            async_copy_b128(&sK[r][off], Kh + g);
            union { v8h v; half_t e[8]; } tv;
            tv.v = *(const v8h*)(V + g);
#pragma unroll
            for (int u = 0; u < 8; ++u) sVt[off + u][r] = tv.e[u];
        }
        wait_async0();
        __syncthreads();

        // ---- sim = q @ k^T  (16 x 64), K-dim = DHEAD in two 32-chunks
        v8f s[4];
#pragma unroll
        for (int fn = 0; fn < 4; ++fn) {
#pragma unroll
            for (int e = 0; e < 8; ++e) s[fn][e] = 0.f;
#pragma unroll
            for (int kc = 0; kc < 2; ++kc) {
                union { v16h v; v8h h2[2]; } bf;
                bf.h2[0] = *(const v8h*)&sK[fn * 16 + m16][kc * 32 + grp * 8];
                bf.h2[1] = *(const v8h*)&sK[fn * 16 + m16][kc * 32 + 16 + grp * 8];
                s[fn] = __builtin_amdgcn_wmma_f32_16x16x32_f16(
                    false, qf[kc].v, false, bf.v, (short)0, s[fn], false, false);
            }
        }

        // ---- online softmax (rows live in 16-lane halves; reduce with shfl_xor)
#pragma unroll
        for (int r = 0; r < 8; ++r) {
            float mx = s[0][r];
#pragma unroll
            for (int fn = 1; fn < 4; ++fn) mx = fmaxf(mx, s[fn][r]);
#pragma unroll
            for (int msk = 8; msk >= 1; msk >>= 1) mx = fmaxf(mx, __shfl_xor(mx, msk, 32));
            const float mnew = fmaxf(mrow[r], mx);
            const float corr = exp2f((mrow[r] - mnew) * 1.44269504f);
            float psum = 0.f;
#pragma unroll
            for (int fn = 0; fn < 4; ++fn) {
                const float p = exp2f((s[fn][r] - mnew) * 1.44269504f);
                s[fn][r] = p;
                psum += p;
            }
#pragma unroll
            for (int msk = 8; msk >= 1; msk >>= 1) psum += __shfl_xor(psum, msk, 32);
            lrow[r] = lrow[r] * corr + psum;
            mrow[r] = mnew;
#pragma unroll
            for (int fd = 0; fd < 4; ++fd) Ofr[fd][r] *= corr;
        }

        // ---- re-materialize P as A-fragments via wave-private LDS
#pragma unroll
        for (int fn = 0; fn < 4; ++fn)
#pragma unroll
            for (int r = 0; r < 8; ++r)
                sP[wv][grp * 8 + r][fn * 16 + m16] = (half_t)s[fn][r];

        // ---- O += P @ V   (K-dim = 64 keys in two 32-chunks)
#pragma unroll
        for (int kc = 0; kc < 2; ++kc) {
            union { v16h v; v8h h2[2]; } pf;
            pf.h2[0] = *(const v8h*)&sP[wv][m16][kc * 32 + grp * 8];
            pf.h2[1] = *(const v8h*)&sP[wv][m16][kc * 32 + 16 + grp * 8];
#pragma unroll
            for (int fd = 0; fd < 4; ++fd) {
                union { v16h v; v8h h2[2]; } vf;
                vf.h2[0] = *(const v8h*)&sVt[fd * 16 + m16][kc * 32 + grp * 8];
                vf.h2[1] = *(const v8h*)&sVt[fd * 16 + m16][kc * 32 + 16 + grp * 8];
                Ofr[fd] = __builtin_amdgcn_wmma_f32_16x16x32_f16(
                    false, pf.v, false, vf.v, (short)0, Ofr[fd], false, false);
            }
        }
        __syncthreads();
    }

    // ---- normalize and store (f16, interleaved back into [B,N,H*D])
#pragma unroll
    for (int fd = 0; fd < 4; ++fd)
#pragma unroll
        for (int r = 0; r < 8; ++r) {
            const size_t row = (size_t)b * SEQ + qRow0 + grp * 8 + r;
            O[row * DMODEL + h * DHEAD + fd * 16 + m16] = (half_t)(Ofr[fd][r] / lrow[r]);
        }
}

// ---------------------------------------------------------------------------
extern "C" void kernel_launch(void* const* d_in, const int* in_sizes, int n_in,
                              void* d_out, int out_size, void* d_ws, size_t ws_size,
                              hipStream_t stream) {
    const float* x   = (const float*)d_in[0];
    const float* ctx = (const float*)d_in[1];
    const float* Wq  = (const float*)d_in[2];
    const float* Wk  = (const float*)d_in[3];
    const float* Wv  = (const float*)d_in[4];
    const float* Wo  = (const float*)d_in[5];
    const float* bo  = (const float*)d_in[6];
    const float* g_q = (const float*)d_in[7];
    const float* b_q = (const float*)d_in[8];
    const float* g_k = (const float*)d_in[9];
    const float* b_k = (const float*)d_in[10];
    const float* g_m = (const float*)d_in[11];
    const float* b_m = (const float*)d_in[12];
    const float* W1  = (const float*)d_in[13];
    const float* b1  = (const float*)d_in[14];
    const float* W2  = (const float*)d_in[15];
    const float* b2  = (const float*)d_in[16];
    float* out = (float*)d_out;

    char* p = (char*)d_ws;
    auto takeH = [&](size_t n) { half_t* r = (half_t*)p; p += n * sizeof(half_t); return r; };
    auto takeF = [&](size_t n) { float*  r = (float*)p;  p += n * sizeof(float);  return r; };

    const size_t RD = (size_t)ROWS * DMODEL;
    half_t* xn  = takeH(RD);
    half_t* cn  = takeH(RD);
    half_t* qh  = takeH(RD);
    half_t* kh  = takeH(RD);
    half_t* vh  = takeH(RD);
    half_t* ao  = takeH(RD);
    half_t* hln = takeH(RD);
    half_t* hid = takeH((size_t)ROWS * HIDDEN);
    float*  x2  = takeF(RD);
    half_t* Wqt = takeH((size_t)DMODEL * DMODEL);
    half_t* Wkt = takeH((size_t)DMODEL * DMODEL);
    half_t* Wvt = takeH((size_t)DMODEL * DMODEL);
    half_t* Wot = takeH((size_t)DMODEL * DMODEL);
    half_t* W1t = takeH((size_t)HIDDEN * DMODEL);
    half_t* W2t = takeH((size_t)DMODEL * HIDDEN);

    const dim3 tB(32, 8);
    transpose_to_f16<<<dim3(DMODEL / 32, DMODEL / 32), tB, 0, stream>>>(Wq, Wqt, DMODEL, DMODEL);
    transpose_to_f16<<<dim3(DMODEL / 32, DMODEL / 32), tB, 0, stream>>>(Wk, Wkt, DMODEL, DMODEL);
    transpose_to_f16<<<dim3(DMODEL / 32, DMODEL / 32), tB, 0, stream>>>(Wv, Wvt, DMODEL, DMODEL);
    transpose_to_f16<<<dim3(DMODEL / 32, DMODEL / 32), tB, 0, stream>>>(Wo, Wot, DMODEL, DMODEL);
    transpose_to_f16<<<dim3(HIDDEN / 32, DMODEL / 32), tB, 0, stream>>>(W1, W1t, DMODEL, HIDDEN);
    transpose_to_f16<<<dim3(DMODEL / 32, HIDDEN / 32), tB, 0, stream>>>(W2, W2t, HIDDEN, DMODEL);

    layernorm_f16<<<ROWS, 256, 0, stream>>>(x,   g_q, b_q, xn);
    layernorm_f16<<<ROWS, 256, 0, stream>>>(ctx, g_k, b_k, cn);

    const dim3 gD(DMODEL / 128, ROWS / 128);   // (8, 64)
    const dim3 gH(HIDDEN / 128, ROWS / 128);   // (32, 64)
    // Q (scale folded), K, V projections -> f16
    gemm_wmma<0><<<gD, 256, 0, stream>>>(xn, Wqt, nullptr, nullptr, nullptr, qh, DMODEL, DMODEL, 0.125f);
    gemm_wmma<0><<<gD, 256, 0, stream>>>(cn, Wkt, nullptr, nullptr, nullptr, kh, DMODEL, DMODEL, 1.0f);
    gemm_wmma<0><<<gD, 256, 0, stream>>>(cn, Wvt, nullptr, nullptr, nullptr, vh, DMODEL, DMODEL, 1.0f);

    attention_wmma<<<dim3(SEQ / 128, B_SZ * NHEADS), 256, 0, stream>>>(qh, kh, vh, ao);

    // x2 = x + attn_out @ Wo + bo
    gemm_wmma<1><<<gD, 256, 0, stream>>>(ao, Wot, bo, x, x2, nullptr, DMODEL, DMODEL, 1.0f);

    layernorm_f16<<<ROWS, 256, 0, stream>>>(x2, g_m, b_m, hln);

    // hid = silu(hln @ W1 + b1)
    gemm_wmma<2><<<gH, 256, 0, stream>>>(hln, W1t, b1, nullptr, nullptr, hid, HIDDEN, DMODEL, 1.0f);
    // out = x2 + hid @ W2 + b2
    gemm_wmma<1><<<gD, 256, 0, stream>>>(hid, W2t, b2, x2, out, nullptr, DMODEL, HIDDEN, 1.0f);
}
